// Encoder_49151605736115
// MI455X (gfx1250) — matrix operand order repaired and checked by reading, not compiled, not run on hardware
//
#include <hip/hip_runtime.h>
#include <hip/hip_bf16.h>

// ---------------- problem constants (match reference) ----------------
#define BB   4
#define SS   512
#define DD   768
#define HH   12
#define NLAYER 2
#define VV   32000
#define EE   8
#define DFFC 3072
#define HDC  64              // DD / HH
#define TT   (BB * SS)       // 2048 tokens
#define CAP  320             // int(1.25 * 2048 / 8)

typedef __attribute__((ext_vector_type(16))) __bf16 v16bf;
typedef __attribute__((ext_vector_type(8)))  float  v8f;

// =====================================================================
// Generic batched GEMM via v_wmma_f32_16x16x32_bf16.
//   C[m,n] = alpha * sum_k A[m,k]*B[k,n]  (+bias[n]) (+relu) (+resid)
//   bt=1: B[k,n] = Bm[n*ldb + k]   (i.e. C = A @ W^T, W row-major [N,K])
//   bt=0: B[k,n] = Bm[k*ldb + n]
// blockIdx.z = batch; decomposed as (zb, zh) = (z/Hdim, z%Hdim) with
// independent element strides so attention (B*H) and experts (E) both map.
// Workgroup: 256 threads / 8 waves; tile 64(M) x 64(N), K-step 32.
// Requires M%64==0, N%64==0, K%32==0 (true for all GEMMs here).
// =====================================================================
__global__ void __launch_bounds__(256)
gemm_bf16_wmma(const float* __restrict__ A, const float* __restrict__ Bm,
               const float* __restrict__ bias, const float* __restrict__ resid,
               float* __restrict__ C,
               int lda, int ldb, int ldc, int M, int N, int K,
               long aBb, long aBh, long bBb, long bBh, long cBb, long cBh,
               long biasStride, int Hdim, float alpha, int bt, int relu)
{
    __shared__ alignas(32) __bf16 aF[4][32][16];   // 4 KB
    __shared__ alignas(32) __bf16 bF[4][32][16];   // 4 KB

    const int tid = threadIdx.x;
    const int z   = blockIdx.z;
    const int zb  = z / Hdim, zh = z % Hdim;
    A  += (long)zb * aBb + (long)zh * aBh;
    Bm += (long)zb * bBb + (long)zh * bBh;
    C  += (long)zb * cBb + (long)zh * cBh;
    if (resid) resid += (long)zb * cBb + (long)zh * cBh;
    if (bias)  bias  += (long)z * biasStride;

    const int m0 = blockIdx.y * 64;
    const int n0 = blockIdx.x * 64;
    const int wave = tid >> 5, lane = tid & 31;
    const int mt = wave >> 1;            // 0..3 : wave's 16-row M tile
    const int np = wave & 1;             // 0..1 : wave's pair of 16-col N tiles

    v8f acc0 = {0.f,0.f,0.f,0.f,0.f,0.f,0.f,0.f};
    v8f acc1 = {0.f,0.f,0.f,0.f,0.f,0.f,0.f,0.f};

    for (int k0 = 0; k0 < K; k0 += 32) {
        // ---- stage A tile: 64 rows x 32 k, f32 -> bf16, fragment order ----
        {
            const int r  = tid >> 3;       // 0..31
            const int f4 = tid & 7;        // float4 index along k
            #pragma unroll
            for (int half = 0; half < 2; ++half) {
                const int row = r + half * 32;
                const float4 v = *(const float4*)&A[(long)(m0 + row) * lda + k0 + f4 * 4];
                const int mrow = row & 15, mtile = row >> 4;
                const float* vp = (const float*)&v;
                #pragma unroll
                for (int j = 0; j < 4; ++j) {
                    const int kk = f4 * 4 + j;
                    // A 16x32 layout (matches B): lanes0-15 K=0..15, lanes16-31 K=16..31
                    const int l = mrow + ((kk >= 16) ? 16 : 0);
                    const int h = kk & 15;
                    aF[mtile][l][h] = (__bf16)vp[j];
                }
            }
        }
        // ---- stage B tile: 32 k x 64 n ----
        if (bt) {           // B[k][n] = Bm[n*ldb + k] : contiguous along k
            const int n  = tid >> 2;       // 0..63
            const int f4 = tid & 3;
            #pragma unroll
            for (int half = 0; half < 2; ++half) {
                const int kq = f4 + half * 4;   // 0..7
                const float4 v = *(const float4*)&Bm[(long)(n0 + n) * ldb + k0 + kq * 4];
                const float* vp = (const float*)&v;
                #pragma unroll
                for (int j = 0; j < 4; ++j) {
                    const int kk = kq * 4 + j;
                    // B 32x16 layout used here (contiguous k, same as A): lanes0-15 K=0..15 (col n), lanes16-31 K=16..31
                    const int l = (n & 15) + ((kk >= 16) ? 16 : 0);
                    bF[n >> 4][l][kk & 15] = (__bf16)vp[j];
                }
            }
        } else {
            // B[k][n] = Bm[k*ldb + n] : thread owns one column n and 8
            // consecutive k inside one 16-group -> loads coalesced across
            // threads (consecutive n) and 8 contiguous LDS halves (pairs
            // into packed cvt + wide ds_store).
            const int n  = tid & 63;       // 0..63
            const int kq = tid >> 6;       // 0..3 -> k = kq*8 .. kq*8+7
            const int l  = (n & 15) + ((kq >= 2) ? 16 : 0);
            const int h0 = (kq * 8) & 15;  // 0 or 8
            const int nt = n >> 4;
            #pragma unroll
            for (int j = 0; j < 8; ++j) {
                const int kk = kq * 8 + j;
                bF[nt][l][h0 + j] = (__bf16)Bm[(long)(k0 + kk) * ldb + n0 + n];
            }
        }
        __syncthreads();

        const v16bf a  = *(const v16bf*)&aF[mt][lane][0];
        const v16bf b0 = *(const v16bf*)&bF[2 * np + 0][lane][0];
        const v16bf b1 = *(const v16bf*)&bF[2 * np + 1][lane][0];
        acc0 = __builtin_amdgcn_wmma_f32_16x16x32_bf16(false, a, false, b0,
                                                       (short)0, acc0, false, false);
        acc1 = __builtin_amdgcn_wmma_f32_16x16x32_bf16(false, a, false, b1,
                                                       (short)0, acc1, false, false);
        __syncthreads();
    }

    // ---- epilogue: C/D layout: lane l, reg r -> M = r + (l>=16?8:0), N = l&15 ----
    const int mBase = m0 + mt * 16 + ((lane >= 16) ? 8 : 0);
    const int nc0   = n0 + (2 * np) * 16 + (lane & 15);
    const int nc1   = nc0 + 16;
    #pragma unroll
    for (int r = 0; r < 8; ++r) {
        const int m = mBase + r;
        float v0 = acc0[r] * alpha;
        float v1 = acc1[r] * alpha;
        if (bias)  { v0 += bias[nc0]; v1 += bias[nc1]; }
        if (relu)  { v0 = fmaxf(v0, 0.f); v1 = fmaxf(v1, 0.f); }
        const long i0 = (long)m * ldc + nc0, i1 = (long)m * ldc + nc1;
        if (resid) { v0 += resid[i0]; v1 += resid[i1]; }
        C[i0] = v0; C[i1] = v1;
    }
}

// =====================================================================
// embedding * sqrt(D) + sinusoidal positional encoding
// =====================================================================
__global__ void __launch_bounds__(256)
embed_pe_k(const int* __restrict__ src, const float* __restrict__ emb,
           float* __restrict__ x)
{
    const int t = blockIdx.x;              // 0..T-1
    const int s = t % SS;
    const long row = (long)src[t] * DD;
    for (int d = threadIdx.x; d < DD; d += 256) {
        const int i2 = d & ~1;
        const float ang = (float)s * __expf(-((float)i2 / (float)DD) * 9.210340371976184f);
        const float pe  = (d & 1) ? __cosf(ang) : __sinf(ang);
        x[(long)t * DD + d] = emb[row + d] * 27.712812921102035f + pe;  // sqrt(768)
    }
}

// =====================================================================
// layernorm over last dim (768), one token per block
// =====================================================================
__global__ void __launch_bounds__(256)
layernorm_k(const float* __restrict__ xin, const float* __restrict__ g,
            const float* __restrict__ b, float* __restrict__ xout)
{
    __shared__ float red[256];
    const int tid = threadIdx.x;
    const float* xr = xin + (long)blockIdx.x * DD;
    float s = 0.f;
    for (int d = tid; d < DD; d += 256) s += xr[d];
    red[tid] = s; __syncthreads();
    for (int o = 128; o > 0; o >>= 1) { if (tid < o) red[tid] += red[tid + o]; __syncthreads(); }
    const float mu = red[0] / (float)DD; __syncthreads();

    float v = 0.f;
    for (int d = tid; d < DD; d += 256) { const float dx = xr[d] - mu; v += dx * dx; }
    red[tid] = v; __syncthreads();
    for (int o = 128; o > 0; o >>= 1) { if (tid < o) red[tid] += red[tid + o]; __syncthreads(); }
    const float rs = rsqrtf(red[0] / (float)DD + 1e-5f);

    float* yr = xout + (long)blockIdx.x * DD;
    for (int d = tid; d < DD; d += 256)
        yr[d] = (xr[d] - mu) * rs * g[d] + b[d];
}

// =====================================================================
// masked softmax over S keys; one (b,h,q) row per block; in place
// =====================================================================
__global__ void __launch_bounds__(256)
attn_softmax_k(float* __restrict__ sc, const unsigned char* __restrict__ pad)
{
    __shared__ float red[256];
    const int tid = threadIdx.x;
    const long row = blockIdx.x;                       // 0 .. B*H*S-1
    const int bidx = (int)(row / (HH * SS));
    float* r = sc + row * SS;

    float mx = -1e30f;
    for (int n = tid; n < SS; n += 256) {
        float v = pad[bidx * SS + n] ? -1e9f : r[n];
        r[n] = v;
        mx = fmaxf(mx, v);
    }
    red[tid] = mx; __syncthreads();
    for (int o = 128; o > 0; o >>= 1) { if (tid < o) red[tid] = fmaxf(red[tid], red[tid + o]); __syncthreads(); }
    mx = red[0]; __syncthreads();

    float sum = 0.f;
    for (int n = tid; n < SS; n += 256) { const float e = __expf(r[n] - mx); r[n] = e; sum += e; }
    red[tid] = sum; __syncthreads();
    for (int o = 128; o > 0; o >>= 1) { if (tid < o) red[tid] += red[tid + o]; __syncthreads(); }
    const float inv = 1.f / red[0];
    for (int n = tid; n < SS; n += 256) r[n] *= inv;
}

// =====================================================================
// router: logits = xn @ wr (N=8 too small for WMMA tiles);
// softmax / argmax / gate / logsumexp, one token per block
// =====================================================================
__global__ void __launch_bounds__(256)
router_k(const float* __restrict__ xn, const float* __restrict__ wr,
         float* __restrict__ probs, float* __restrict__ gate,
         int* __restrict__ eidx, float* __restrict__ lse)
{
    __shared__ float sh[256 * EE];
    const int tid = threadIdx.x;
    const int t = blockIdx.x;
    const float* xr = xn + (long)t * DD;

    float acc[EE];
    #pragma unroll
    for (int e = 0; e < EE; ++e) acc[e] = 0.f;
    for (int d = tid; d < DD; d += 256) {
        const float xv = xr[d];
        const float* wrow = wr + (long)d * EE;
        #pragma unroll
        for (int e = 0; e < EE; ++e) acc[e] += xv * wrow[e];
    }
    #pragma unroll
    for (int e = 0; e < EE; ++e) sh[tid * EE + e] = acc[e];
    __syncthreads();
    for (int o = 128; o > 0; o >>= 1) {
        if (tid < o) {
            #pragma unroll
            for (int e = 0; e < EE; ++e) sh[tid * EE + e] += sh[(tid + o) * EE + e];
        }
        __syncthreads();
    }
    if (tid == 0) {
        float lg[EE], p[EE];
        #pragma unroll
        for (int e = 0; e < EE; ++e) lg[e] = sh[e];
        float mx = lg[0]; int am = 0;
        #pragma unroll
        for (int e = 1; e < EE; ++e) { if (lg[e] > mx) mx = lg[e]; if (lg[e] > lg[am]) am = e; }
        float sum = 0.f;
        #pragma unroll
        for (int e = 0; e < EE; ++e) { p[e] = __expf(lg[e] - mx); sum += p[e]; }
        const float inv = 1.f / sum;
        #pragma unroll
        for (int e = 0; e < EE; ++e) probs[(long)t * EE + e] = p[e] * inv;
        gate[t] = p[am] * inv;
        eidx[t] = am;
        lse[t]  = mx + __logf(sum);
    }
}

// =====================================================================
// top-1 capacity dispatch: sequential cumsum over tokens (order matters),
// plus aux losses (load-balance lb and router z-loss), accumulated /NL.
// =====================================================================
__global__ void __launch_bounds__(256)
dispatch_k(const unsigned char* __restrict__ pad, const float* __restrict__ probs,
           const int* __restrict__ eidx, const float* __restrict__ lse,
           int* __restrict__ tos, float* __restrict__ lossA)
{
    const int tid = threadIdx.x;
    for (int i = tid; i < EE * CAP; i += 256) tos[i] = -1;
    __syncthreads();
    if (tid == 0) {
        int   oh[EE];  float me[EE];
        #pragma unroll
        for (int e = 0; e < EE; ++e) { oh[e] = 0; me[e] = 0.f; }
        float zsum = 0.f, denom = 0.f;
        for (int t = 0; t < TT; ++t) {
            if (!pad[t]) {
                denom += 1.f;
                const float* pr = probs + (long)t * EE;
                #pragma unroll
                for (int e = 0; e < EE; ++e) me[e] += pr[e];
                const int e = eidx[t];
                oh[e] += 1;
                if (oh[e] <= CAP) tos[e * CAP + oh[e] - 1] = t;
                const float l = lse[t];
                zsum += l * l;
            }
        }
        denom = fmaxf(denom, 1.f);
        float lb = 0.f;
        #pragma unroll
        for (int e = 0; e < EE; ++e) lb += (me[e] / denom) * ((float)oh[e] / denom);
        lb *= (float)EE;
        atomicAdd(&lossA[0], lb / (float)NLAYER);
        atomicAdd(&lossA[1], (zsum / denom) / (float)NLAYER);
    }
}

// gather dispatched tokens into expert slots; zero empty slots
__global__ void __launch_bounds__(256)
gather_k(const float* __restrict__ xn, const int* __restrict__ tos,
         float* __restrict__ ein)
{
    const int slot = blockIdx.x;                 // 0 .. E*CAP-1
    const int t = tos[slot];
    float* dst = ein + (long)slot * DD;
    if (t < 0) {
        for (int d = threadIdx.x; d < DD; d += 256) dst[d] = 0.f;
    } else {
        const float* srcp = xn + (long)t * DD;
        for (int d = threadIdx.x; d < DD; d += 256) dst[d] = srcp[d];
    }
}

// scatter expert outputs back (x += gate * eo); top-1 => no conflicts
__global__ void __launch_bounds__(256)
scatter_k(const int* __restrict__ tos, const float* __restrict__ gate,
          const float* __restrict__ eo, float* __restrict__ x)
{
    const int slot = blockIdx.x;
    const int t = tos[slot];
    if (t < 0) return;
    const float gsc = gate[t];
    const float* src = eo + (long)slot * DD;
    float* dst = x + (long)t * DD;
    for (int d = threadIdx.x; d < DD; d += 256) dst[d] += gsc * src[d];
}

__global__ void zero_loss_k(float* lossA) {
    if (threadIdx.x < 2) lossA[threadIdx.x] = 0.f;
}

__global__ void write_loss_k(const float* __restrict__ lossA, float* __restrict__ out) {
    if (threadIdx.x == 0) { out[0] = lossA[0]; out[1] = lossA[1]; }
}

// =====================================================================
extern "C" void kernel_launch(void* const* d_in, const int* in_sizes, int n_in,
                              void* d_out, int out_size, void* d_ws, size_t ws_size,
                              hipStream_t stream)
{
    (void)in_sizes; (void)n_in; (void)out_size; (void)ws_size;
    const int*           src   = (const int*)d_in[0];
    const unsigned char* pad   = (const unsigned char*)d_in[1];  // all-false mask
    const float* emb   = (const float*)d_in[2];
    const float* qkv_w = (const float*)d_in[3];
    const float* qkv_b = (const float*)d_in[4];
    const float* out_w = (const float*)d_in[5];
    const float* out_b = (const float*)d_in[6];
    const float* wr    = (const float*)d_in[7];
    const float* w1    = (const float*)d_in[8];
    const float* b1    = (const float*)d_in[9];
    const float* w2    = (const float*)d_in[10];
    const float* b2    = (const float*)d_in[11];
    const float* ln1g  = (const float*)d_in[12];
    const float* ln1b  = (const float*)d_in[13];
    const float* ln2g  = (const float*)d_in[14];
    const float* ln2b  = (const float*)d_in[15];
    const float* lnfg  = (const float*)d_in[16];
    const float* lnfb  = (const float*)d_in[17];
    float* out = (float*)d_out;

    // ---- workspace carve-up (floats) ----
    float* ws = (float*)d_ws;
    size_t o = 0;
    float* x     = ws + o; o += (size_t)TT * DD;         // 1.57M
    float* xn    = ws + o; o += (size_t)TT * DD;
    float* qkv   = ws + o; o += (size_t)TT * 3 * DD;     // 4.72M
    float* ao    = ws + o; o += (size_t)TT * DD;         // attention out
    float* probs = ws + o; o += (size_t)TT * EE;
    float* gate  = ws + o; o += TT;
    float* lse   = ws + o; o += TT;
    float* lossA = ws + o; o += 8;
    int*   tos   = (int*)(ws + o); o += (size_t)EE * CAP;
    int*   eidx  = (int*)(ws + o); o += TT;
    float* scores = ws + o; o += (size_t)BB * HH * SS * SS;   // 12.58M, reused:
    float* ein = scores;                                  // E*CAP*D   = 1.97M
    float* hid = ein + (size_t)EE * CAP * DD;             // E*CAP*DFF = 7.86M
    float* eo  = hid + (size_t)EE * CAP * DFFC;           // E*CAP*D   = 1.97M

    const dim3 blk(256);
    auto gemm = [&](const float* A, const float* Bm, const float* bias,
                    const float* resid, float* C,
                    int lda, int ldb, int ldc, int M, int N, int K,
                    long aBb, long aBh, long bBb, long bBh, long cBb, long cBh,
                    long biasStride, int Hdim, int nb, float alpha, int bt, int relu) {
        dim3 grid(N / 64, M / 64, nb);
        gemm_bf16_wmma<<<grid, blk, 0, stream>>>(A, Bm, bias, resid, C,
            lda, ldb, ldc, M, N, K, aBb, aBh, bBb, bBh, cBb, cBh,
            biasStride, Hdim, alpha, bt, relu);
    };

    zero_loss_k<<<1, 32, 0, stream>>>(lossA);
    embed_pe_k<<<TT, blk, 0, stream>>>(src, emb, x);

    for (int l = 0; l < NLAYER; ++l) {
        const float* qkvW = qkv_w + (long)l * 3 * DD * DD;
        const float* qkvB = qkv_b + (long)l * 3 * DD;
        const float* outW = out_w + (long)l * DD * DD;
        const float* outB = out_b + (long)l * DD;
        const float* wrL  = wr    + (long)l * DD * EE;
        const float* w1L  = w1    + (long)l * EE * DD * DFFC;
        const float* b1L  = b1    + (long)l * EE * DFFC;
        const float* w2L  = w2    + (long)l * EE * DFFC * DD;
        const float* b2L  = b2    + (long)l * EE * DD;

        // ---- attention ----
        layernorm_k<<<TT, blk, 0, stream>>>(x, ln1g + l * DD, ln1b + l * DD, xn);

        // qkv = xn @ qkvW^T + qkvB : [2048 x 2304]
        gemm(xn, qkvW, qkvB, nullptr, qkv,
             DD, DD, 3 * DD, TT, 3 * DD, DD,
             0, 0, 0, 0, 0, 0, 0, 1, 1, 1.0f, /*bt=*/1, 0);

        // scores = q @ k^T / sqrt(HD) : batched over B*H, [512 x 512], K=64
        gemm(qkv /*q*/, qkv + DD /*k*/, nullptr, nullptr, scores,
             3 * DD, 3 * DD, SS, SS, SS, HDC,
             (long)SS * 3 * DD, HDC,               // A: b, h strides
             (long)SS * 3 * DD, HDC,               // B: b, h strides
             (long)HH * SS * SS, (long)SS * SS,    // C: b, h strides
             0, HH, BB * HH, 0.125f, /*bt=*/1, 0);

        attn_softmax_k<<<BB * HH * SS, blk, 0, stream>>>(scores, pad);

        // ao = attn @ v : batched [512 x 64], K=512
        gemm(scores, qkv + 2 * DD /*v*/, nullptr, nullptr, ao,
             SS, 3 * DD, DD, SS, HDC, SS,
             (long)HH * SS * SS, (long)SS * SS,
             (long)SS * 3 * DD, HDC,
             (long)SS * DD, HDC,
             0, HH, BB * HH, 1.0f, /*bt=*/0, 0);

        // x = x + ao @ outW^T + outB
        gemm(ao, outW, outB, x /*resid*/, x,
             DD, DD, DD, TT, DD, DD,
             0, 0, 0, 0, 0, 0, 0, 1, 1, 1.0f, /*bt=*/1, 0);

        // ---- switch FFN ----
        layernorm_k<<<TT, blk, 0, stream>>>(x, ln2g + l * DD, ln2b + l * DD, xn);
        router_k<<<TT, blk, 0, stream>>>(xn, wrL, probs, gate, eidx, lse);
        dispatch_k<<<1, blk, 0, stream>>>(pad, probs, eidx, lse, tos, lossA);
        gather_k<<<EE * CAP, blk, 0, stream>>>(xn, tos, ein);

        // hid = relu(ein @ w1[e] + b1[e]) : batched over E, [320 x 3072], K=768
        gemm(ein, w1L, b1L, nullptr, hid,
             DD, DFFC, DFFC, CAP, DFFC, DD,
             (long)CAP * DD, 0,
             (long)DD * DFFC, 0,
             (long)CAP * DFFC, 0,
             DFFC, 1, EE, 1.0f, /*bt=*/0, /*relu=*/1);

        // eo = hid @ w2[e] + b2[e] : batched over E, [320 x 768], K=3072
        gemm(hid, w2L, b2L, nullptr, eo,
             DFFC, DD, DD, CAP, DD, DFFC,
             (long)CAP * DFFC, 0,
             (long)DFFC * DD, 0,
             (long)CAP * DD, 0,
             DD, 1, EE, 1.0f, /*bt=*/0, 0);

        scatter_k<<<EE * CAP, blk, 0, stream>>>(tos, gate, eo, x);
    }

    // final layernorm -> d_out[0 : T*D], then the two scalar aux losses
    layernorm_k<<<TT, blk, 0, stream>>>(x, lnfg, lnfb, out);
    write_loss_k<<<1, 32, 0, stream>>>(lossA, out + (size_t)TT * DD);
}
